// DecentralizedControlNet_32143535243486
// MI455X (gfx1250) — matrix operand order repaired and checked by reading, not compiled, not run on hardware
//
#include <hip/hip_runtime.h>

// ---------------------------------------------------------------------------
// DecentralizedControlNet on MI455X (gfx1250, wave32, WMMA)
//
// Per block (256 thr = 8 waves, 32 agents):
//  1. LDS copy of edge-padded error field (4422 f32)
//  2. local obs: 327-window + gradient + antialiased bilinear -> [32x64] bf16
//  3. WMMA bf16 16x16x32: patches @ W0t -> +bias -> GN(1)+tanh   [32x256]
//  4. WMMA:               h @ W1t       -> +bias -> GN(1)+tanh   [32x256]
//     trunk net (scalar) -> cols 256..287 of same buffer
//  5. WMMA:               [h|enc] @ cWt -> f32 [32x32]
//  6. heads: tanh(+cb), 32-dots with uW/vW, 40*tanh / tanh -> d_out
//
// GN+tanh runs on all 256 threads (8 threads/row, __shfl_xor reduction);
// hidden tanh uses hardware v_exp_f32. Weights transposed to bf16 [N][K]
// in d_ws so each B-fragment is two 16B row loads (L2/WGP$-resident).
// ---------------------------------------------------------------------------

typedef __attribute__((ext_vector_type(16))) __bf16 v16bf;
typedef __attribute__((ext_vector_type(8)))  __bf16 v8bf;
typedef __attribute__((ext_vector_type(8)))  float  v8f;

#define N_PDE     4096
#define WINDOW    327
#define HALF      163
#define PAD_LEN   (N_PDE + 2 * HALF)   // 4422
#define AGB       32                   // agents per block
#define SCALE     16.35f               // 327/20
#define INV_SCALE (1.0f / 16.35f)

// ws layout (bytes)
#define W0T_OFF   0            // [256][64]  bf16 = 32768
#define W1T_OFF   32768        // [256][256] bf16 = 131072
#define CWT_OFF   163840       // [32][288]  bf16 = 18432
#define PREP_N    (256 * 64 + 256 * 256 + 32 * 288)   // 91136

// LDS layout (bytes)
#define ERR_OFF   0                    // 4422 f32 = 17688 (aliased later: xbuf)
#define PAT_OFF   17696                // 32*64  bf16 = 4096
#define HA_OFF    21792                // 32*256 bf16 = 16384
#define HB_OFF    38176                // 32*288 bf16 = 18432
#define SMEM_SIZE 56608

// ---------------------------------------------------------------------------
__device__ __forceinline__ float fast_tanh(float x) {
  // bf16-accurate tanh: one v_exp_f32 + fast divide
  x = fminf(fmaxf(x, -10.0f), 10.0f);
  float e = __expf(2.0f * x);
  return __fdividef(e - 1.0f, e + 1.0f);
}

// ---------------------------------------------------------------------------
// Prep: transpose weights into bf16 [N][K] panels in scratch.
// ---------------------------------------------------------------------------
__global__ void __launch_bounds__(256) dcn_prep(
    const float* __restrict__ bW0,   // [40][256]
    const float* __restrict__ bW1,   // [256][256]
    const float* __restrict__ cW,    // [288][32]
    char* __restrict__ ws)
{
  int idx = blockIdx.x * 256 + threadIdx.x;
  __bf16* w0t = (__bf16*)(ws + W0T_OFF);
  __bf16* w1t = (__bf16*)(ws + W1T_OFF);
  __bf16* cwt = (__bf16*)(ws + CWT_OFF);
  if (idx < 256 * 64) {                       // W0t[n][k], K padded to 64
    int n = idx >> 6, k = idx & 63;
    float v = (k < 40) ? bW0[k * 256 + n] : 0.0f;
    w0t[n * 64 + k] = (__bf16)v;
  } else if (idx < 256 * 64 + 256 * 256) {    // W1t[n][k]
    int i = idx - 256 * 64;
    int n = i >> 8, k = i & 255;
    w1t[n * 256 + k] = (__bf16)bW1[k * 256 + n];
  } else if (idx < PREP_N) {                  // cWt[n][k], K=288
    int i = idx - 256 * 64 - 256 * 256;
    int n = i / 288, k = i % 288;
    cwt[n * 288 + k] = (__bf16)cW[k * 32 + n];
  }
}

// ---------------------------------------------------------------------------
// WMMA tile layer: act[ngroups*16 x K] (LDS bf16) @ wt[N x K] (global bf16,
// transposed) -> out (+bias), out bf16 (layers) or f32 (head input).
// ---------------------------------------------------------------------------
template <bool F32OUT>
__device__ __forceinline__ void wmma_layer(
    const __bf16* __restrict__ act, int strideA,
    const __bf16* __restrict__ wt,  int strideW,
    void* outBuf, int strideO,
    const float* __restrict__ bias,
    int ngroups, int ntiles, int kchunks)
{
  const int wv   = threadIdx.x >> 5;
  const int lane = threadIdx.x & 31;
  const int half = lane >> 4;
  const int l16  = lane & 15;
  const int njobs = ngroups * ntiles;

  for (int job = wv; job < njobs; job += 8) {
    const int g  = job / ntiles;
    const int nt = job % ntiles;
    v8f acc = {};
    // A: row = agent (M); elems 0..7 = K kc*32+half*8.., elems 8..15 = +16
    const __bf16* arow = act + (g * 16 + l16) * strideA + half * 8;
    // B: row = output col (N); elems 0..7 = K kc*32+half*8.., elems 8..15 = +16
    const __bf16* brow = wt + (nt * 16 + l16) * strideW + half * 8;
    for (int kc = 0; kc < kchunks; ++kc) {
      v8bf a0 = *(const v8bf*)(arow + kc * 32);
      v8bf a1 = *(const v8bf*)(arow + kc * 32 + 16);
      v8bf b0 = *(const v8bf*)(brow + kc * 32);
      v8bf b1 = *(const v8bf*)(brow + kc * 32 + 16);
      v16bf a = __builtin_shufflevector(a0, a1, 0,1,2,3,4,5,6,7,8,9,10,11,12,13,14,15);
      v16bf b = __builtin_shufflevector(b0, b1, 0,1,2,3,4,5,6,7,8,9,10,11,12,13,14,15);
      acc = __builtin_amdgcn_wmma_f32_16x16x32_bf16(
          false, a, false, b, (short)0, acc, false, false);
    }
    // D layout: lanes 0-15 -> M=r, lanes 16-31 -> M=8+r; N = lane&15
    const int mbase = half * 8;
    const int col   = nt * 16 + l16;
    const float bv  = bias ? bias[col] : 0.0f;
#pragma unroll
    for (int r = 0; r < 8; ++r) {
      int row = g * 16 + mbase + r;
      float v = acc[r] + bv;
      if (F32OUT) ((float*)outBuf)[row * strideO + col] = v;
      else        ((__bf16*)outBuf)[row * strideO + col] = (__bf16)v;
    }
  }
}

// ---------------------------------------------------------------------------
// GroupNorm(1 group, 256 feats) + tanh, all 256 threads: 8 threads per row,
// 32 features each; cross-thread reduction via wave32 __shfl_xor (masks 1/2/4
// stay inside the 8-lane group).
// ---------------------------------------------------------------------------
__device__ __forceinline__ void gn_tanh_par(
    __bf16* buf, int stride,
    const float* __restrict__ sc, const float* __restrict__ bi)
{
  const int t   = threadIdx.x;
  const int row = t >> 3;
  const int seg = t & 7;
  __bf16* r = buf + row * stride + seg * 32;

  float s = 0.0f, s2 = 0.0f;
#pragma unroll
  for (int c = 0; c < 4; ++c) {            // 4 x v8bf = 32 elements
    v8bf v = *(const v8bf*)(r + c * 8);
#pragma unroll
    for (int i = 0; i < 8; ++i) {
      float x = (float)v[i];
      s += x; s2 += x * x;
    }
  }
  s  += __shfl_xor(s, 1, 32);  s2 += __shfl_xor(s2, 1, 32);
  s  += __shfl_xor(s, 2, 32);  s2 += __shfl_xor(s2, 2, 32);
  s  += __shfl_xor(s, 4, 32);  s2 += __shfl_xor(s2, 4, 32);

  float mu  = s * (1.0f / 256.0f);
  float var = s2 * (1.0f / 256.0f) - mu * mu;
  float rs  = rsqrtf(var + 1e-6f);

#pragma unroll
  for (int c = 0; c < 4; ++c) {
    v8bf v = *(const v8bf*)(r + c * 8);
    v8bf o;
#pragma unroll
    for (int i = 0; i < 8; ++i) {
      int col = seg * 32 + c * 8 + i;
      float x = (float)v[i];
      o[i] = (__bf16)fast_tanh((x - mu) * rs * sc[col] + bi[col]);
    }
    *(v8bf*)(r + c * 8) = o;
  }
}

// ---------------------------------------------------------------------------
// Main fused kernel
// ---------------------------------------------------------------------------
__global__ void __launch_bounds__(256) dcn_kernel(
    const float* __restrict__ z_curr, const float* __restrict__ z_target,
    const float* __restrict__ xi_curr,
    const float* __restrict__ bb0, const float* __restrict__ bs0, const float* __restrict__ bB0,
    const float* __restrict__ bb1, const float* __restrict__ bs1, const float* __restrict__ bB1,
    const float* __restrict__ tW0, const float* __restrict__ tb0,
    const float* __restrict__ tW1, const float* __restrict__ tb1,
    const float* __restrict__ cb,
    const float* __restrict__ uW,  const float* __restrict__ ubp,
    const float* __restrict__ vW,  const float* __restrict__ vbp,
    const char* __restrict__ ws, float* __restrict__ out, int nA)
{
  __shared__ char smem[SMEM_SIZE];
  float*  errS = (float*)(smem + ERR_OFF);   // 4422 f32
  __bf16* pat  = (__bf16*)(smem + PAT_OFF);  // [32][64]
  __bf16* hA   = (__bf16*)(smem + HA_OFF);   // [32][256]
  __bf16* hB   = (__bf16*)(smem + HB_OFF);   // [32][288] (cols 256..287 = enc)
  float*  xbuf = (float*)(smem + ERR_OFF);   // alias, [32][32] f32

  const int tid    = threadIdx.x;
  const int block0 = blockIdx.x * AGB;

  const __bf16* w0t = (const __bf16*)(ws + W0T_OFF);
  const __bf16* w1t = (const __bf16*)(ws + W1T_OFF);
  const __bf16* cwt = (const __bf16*)(ws + CWT_OFF);

  // warm L2/WGP$ for the big weight panel (global_prefetch_b8)
  for (int o = tid * 256; o < 131072; o += 65536)
    __builtin_prefetch(ws + W1T_OFF + o, 0, 0);

  // --- 1. edge-padded error field into LDS ------------------------------
  for (int i = tid; i < PAD_LEN; i += 256) {
    int c = i - HALF;
    c = c < 0 ? 0 : (c > N_PDE - 1 ? N_PDE - 1 : c);
    errS[i] = z_curr[c] - z_target[c];
  }
  __syncthreads();

  // --- 2. patches: antialiased bilinear 327->20 of err + gradient -------
  {
    const int a    = tid >> 3;     // agent 0..31
    const int part = tid & 7;      // 8 threads per agent
    const float xi = xi_curr[block0 + a];
    const int start = (int)(xi * (float)(N_PDE - 1));   // trunc, xi >= 0

    for (int j = part; j < 20; j += 8) {
      const float cc = (j + 0.5f) * SCALE - 0.5f;
      int lo = (int)ceilf(cc - SCALE);  if (lo < 0)   lo = 0;
      int hi = (int)floorf(cc + SCALE); if (hi > 326) hi = 326;
      float se = 0.0f, sg = 0.0f, swt = 0.0f;
      for (int k = lo; k <= hi; ++k) {
        float wt = 1.0f - fabsf((float)k - cc) * INV_SCALE;
        if (wt <= 0.0f) continue;
        float e = errS[start + k];
        float g;
        if (k == 0)        g = errS[start + 1]   - errS[start];
        else if (k == 326) g = errS[start + 326] - errS[start + 325];
        else               g = 0.5f * (errS[start + k + 1] - errS[start + k - 1]);
        se += wt * e; sg += wt * g; swt += wt;
      }
      float inv = 1.0f / swt;
      pat[a * 64 + j]      = (__bf16)(se * inv);
      pat[a * 64 + 20 + j] = (__bf16)(sg * inv);
    }
    for (int c = 40 + part; c < 64; c += 8)   // zero K-pad 40..63
      pat[a * 64 + c] = (__bf16)0.0f;
  }
  __syncthreads();

  // --- 3. branch layer 0: [32x64] @ [64x256] ----------------------------
  wmma_layer<false>(pat, 64, w0t, 64, hA, 256, bb0, AGB / 16, 16, 2);
  __syncthreads();
  gn_tanh_par(hA, 256, bs0, bB0);
  __syncthreads();

  // --- 4. branch layer 1: [32x256] @ [256x256] --------------------------
  wmma_layer<false>(hA, 256, w1t, 256, hB, 288, bb1, AGB / 16, 16, 8);
  __syncthreads();
  gn_tanh_par(hB, 288, bs1, bB1);          // cols 0..255, all threads
  if (tid < AGB) {                          // trunk -> cols 256..287 (disjoint)
    const int a = tid;
    const float xi = xi_curr[block0 + a];
    const float PI = 3.14159265358979f;
    float enc[8];
    const float fr[4] = {1.0f, 2.0f, 4.0f, 8.0f};
#pragma unroll
    for (int f = 0; f < 4; ++f) {
      float ang = xi * fr[f] * PI;
      enc[f] = __sinf(ang); enc[4 + f] = __cosf(ang);
    }
    float t1[32];
    for (int n = 0; n < 32; ++n) {
      float s = tb0[n];
#pragma unroll
      for (int k = 0; k < 8; ++k) s += enc[k] * tW0[k * 32 + n];
      t1[n] = fast_tanh(s);
    }
    for (int n = 0; n < 32; ++n) {
      float s = tb1[n];
      for (int k = 0; k < 32; ++k) s += t1[k] * tW1[k * 32 + n];
      hB[a * 288 + 256 + n] = (__bf16)fast_tanh(s);
    }
  }
  __syncthreads();

  // --- 5. combined head: [32x288] @ [288x32] -> f32 ---------------------
  wmma_layer<true>(hB, 288, cwt, 288, xbuf, 32, nullptr, AGB / 16, 2, 9);
  __syncthreads();

  // --- 6. output heads (full-precision tanh on final outputs) -----------
  if (tid < AGB) {
    float x[32];
    for (int n = 0; n < 32; ++n) x[n] = tanhf(xbuf[tid * 32 + n] + cb[n]);
    float u = ubp[0], v = vbp[0];
    for (int k = 0; k < 32; ++k) { u += x[k] * uW[k]; v += x[k] * vW[k]; }
    const int a = block0 + tid;
    out[a]      = 40.0f * tanhf(u);
    out[nA + a] = tanhf(v);
  }
}

// ---------------------------------------------------------------------------
extern "C" void kernel_launch(void* const* d_in, const int* in_sizes, int n_in,
                              void* d_out, int out_size, void* d_ws, size_t ws_size,
                              hipStream_t stream) {
  const float* z_curr   = (const float*)d_in[0];
  const float* z_target = (const float*)d_in[1];
  const float* xi_curr  = (const float*)d_in[2];
  const float* bW0 = (const float*)d_in[3];
  const float* bb0 = (const float*)d_in[4];
  const float* bs0 = (const float*)d_in[5];
  const float* bB0 = (const float*)d_in[6];
  const float* bW1 = (const float*)d_in[7];
  const float* bb1 = (const float*)d_in[8];
  const float* bs1 = (const float*)d_in[9];
  const float* bB1 = (const float*)d_in[10];
  const float* tW0 = (const float*)d_in[11];
  const float* tb0 = (const float*)d_in[12];
  const float* tW1 = (const float*)d_in[13];
  const float* tb1 = (const float*)d_in[14];
  const float* cW  = (const float*)d_in[15];
  const float* cb  = (const float*)d_in[16];
  const float* uW  = (const float*)d_in[17];
  const float* ub  = (const float*)d_in[18];
  const float* vW  = (const float*)d_in[19];
  const float* vb  = (const float*)d_in[20];

  const int nA = in_sizes[2];
  char* ws = (char*)d_ws;

  dcn_prep<<<(PREP_N + 255) / 256, 256, 0, stream>>>(bW0, bW1, cW, ws);

  dcn_kernel<<<nA / AGB, 256, 0, stream>>>(
      z_curr, z_target, xi_curr,
      bb0, bs0, bB0, bb1, bs1, bB1,
      tW0, tb0, tW1, tb1, cb, uW, ub, vW, vb,
      ws, (float*)d_out, nA);
}
